// MixtureOfExperts_16131897164150
// MI455X (gfx1250) — compile-verified, matrix operand order checked by reading, not run on hardware
//
#include <hip/hip_runtime.h>
#include <hip/hip_bf16.h>
#include <stdint.h>

// ---------------- types for WMMA ----------------
typedef __bf16  v16bf __attribute__((ext_vector_type(16)));
typedef float   v8f   __attribute__((ext_vector_type(8)));
typedef unsigned short ushort8  __attribute__((ext_vector_type(8)));
typedef unsigned short ushort16 __attribute__((ext_vector_type(16)));

// ---------------- constants ----------------
#define BATCH   65536
#define DDIM    256
#define GDIM    128
#define EEXP    8
#define H0DIM   256
#define H1DIM   128
#define H2DIM   64
#define EPSBN   1e-5f

// LDS per-wave layout (elements), padded strides to avoid bank conflicts
#define XN_LD   264   // bf16 stride for 256-wide
#define H0_LD   264
#define H1_LD   136   // bf16 stride for 128-wide
#define RED_LD  68    // f32 stride for 64-wide
#define WAVE_BYTES (16*XN_LD*2 + 16*H0_LD*2 + 16*H1_LD*2 + 16*8*4 + 16*RED_LD*4) // 26112
#define NWAVES  4
// block-shared region: gW2 [128][8] f32 + gb2 [8] f32 + oW [64] f32
#define BLK_BYTES (GDIM*EEXP*4 + EEXP*4 + H2DIM*4)        // 4384
#define SMEM_BYTES (WAVE_BYTES * NWAVES + BLK_BYTES)      // 108832 < 320KB/WG

// ---------------- helpers ----------------
static __device__ __forceinline__ unsigned short f2bf(float f) {
    unsigned int u = __builtin_bit_cast(unsigned int, f);
    u += 0x7FFFu + ((u >> 16) & 1u);     // round-to-nearest-even
    return (unsigned short)(u >> 16);
}
static __device__ __forceinline__ float bf2f(unsigned short h) {
    unsigned int u = ((unsigned int)h) << 16;
    return __builtin_bit_cast(float, u);
}
// Load one 16-element bf16 fragment: 8 bf16 at p[0..7], 8 bf16 at p[16..23]
static __device__ __forceinline__ v16bf frag_ld(const unsigned short* p) {
    ushort8 lo = *(const ushort8*)(p);
    ushort8 hi = *(const ushort8*)(p + 16);
    ushort16 w = __builtin_shufflevector(lo, hi, 0,1,2,3,4,5,6,7,8,9,10,11,12,13,14,15);
    return __builtin_bit_cast(v16bf, w);
}

// GEMM: C[16,N] = relu(A[16,K] * Wt[N,K]^T + bias), output as bf16 into LDS
static __device__ __forceinline__ void gemm_bias_relu_bf16(
    const unsigned short* __restrict__ A, int lda,      // LDS, [16][lda] bf16
    const unsigned short* __restrict__ Wt,              // global, [N][K] bf16 (BN-folded)
    int K, int N,
    const float* __restrict__ bias,                     // global, [N] f32 (BN-folded)
    unsigned short* __restrict__ Out, int ldo,          // LDS, [16][ldo] bf16
    int mr, int kb, int half)
{
    const unsigned short* arow = A + mr * lda + kb;
    for (int nt = 0; nt < (N >> 4); ++nt) {
        const int ncol = nt * 16 + mr;
        const unsigned short* wrow = Wt + (size_t)ncol * K + kb;
        v8f c = {};
        for (int k0 = 0; k0 < K; k0 += 32) {
            v16bf a = frag_ld(arow + k0);
            v16bf b = frag_ld(wrow + k0);
            c = __builtin_amdgcn_wmma_f32_16x16x32_bf16(false, a, false, b,
                                                        (short)0, c, false, false);
        }
        const float bv = bias[ncol];
#pragma unroll
        for (int r = 0; r < 8; ++r) {
            int m = half ? (r + 8) : r;
            Out[m * ldo + ncol] = f2bf(fmaxf(c[r] + bv, 0.f));   // v_max_num_f32
        }
    }
}

// ---------------- preprocessing kernels ----------------
__global__ void prep_in_bn(const float* __restrict__ g, const float* __restrict__ b,
                           const float* __restrict__ m, const float* __restrict__ v,
                           float* __restrict__ s, float* __restrict__ t) {
    int i = blockIdx.x * blockDim.x + threadIdx.x;
    if (i < DDIM) {
        float sc = g[i] * rsqrtf(v[i] + EPSBN);
        s[i] = sc;
        t[i] = b[i] - m[i] * sc;
    }
}

// Fold BN into Linear, transpose [E][K][N] -> [E][N][K], convert to bf16.
__global__ void prep_fold(const float* __restrict__ W, const float* __restrict__ bias,
                          const float* __restrict__ gg, const float* __restrict__ bb,
                          const float* __restrict__ mm, const float* __restrict__ vv,
                          unsigned short* __restrict__ Wt, float* __restrict__ bout,
                          int K, int N, int total) {
    int idx = blockIdx.x * blockDim.x + threadIdx.x;
    if (idx >= total) return;
    int k = idx % K;
    int rem = idx / K;
    int n = rem % N;
    int e = rem / N;
    int en = e * N + n;
    float sc = gg[en] * rsqrtf(vv[en] + EPSBN);
    Wt[idx] = f2bf(W[(size_t)e * K * N + (size_t)k * N + n] * sc);
    if (k == 0) bout[en] = (bias[en] - mm[en]) * sc + bb[en];
}

// ---------------- fused MoE kernel ----------------
// 128 threads = 4 waves; each wave owns 16 rows end-to-end (no cross-wave LDS deps
// after the one stage-in barrier; LDS ops from one wave retire in order).
__global__ void __launch_bounds__(128)
moe_fused(const float* __restrict__ x,
          const float* __restrict__ in_s, const float* __restrict__ in_t,
          const unsigned short* __restrict__ g1t, const float* __restrict__ g1b,
          const float* __restrict__ gW2, const float* __restrict__ gb2,
          const unsigned short* __restrict__ e0t, const float* __restrict__ e0b,
          const unsigned short* __restrict__ e1t, const float* __restrict__ e1b,
          const unsigned short* __restrict__ e2t, const float* __restrict__ e2b,
          const float* __restrict__ oW, const float* __restrict__ ob,
          float* __restrict__ out)
{
    extern __shared__ __align__(16) char smem[];
    const int tid  = threadIdx.x;
    const int wave = tid >> 5;
    const int lane = tid & 31;
    const int half = lane >> 4;
    const int mr   = lane & 15;
    const int kb   = half * 8;

    char* wbase = smem + wave * WAVE_BYTES;
    unsigned short* XN  = (unsigned short*)wbase;          // [16][XN_LD] bf16
    unsigned short* H0B = XN  + 16 * XN_LD;                // [16][H0_LD] bf16
    unsigned short* H1B = H0B + 16 * H0_LD;                // [16][H1_LD] bf16
    float* GATES = (float*)(H1B + 16 * H1_LD);             // [16][8]
    float* RED   = GATES + 16 * 8;                         // [16][RED_LD]

    // block-shared small tensors
    float* GW2S = (float*)(smem + WAVE_BYTES * NWAVES);    // [128][8]
    float* GB2S = GW2S + GDIM * EEXP;                      // [8]
    float* OWS  = GB2S + EEXP;                             // [64]

    // cooperative stage-in of tiny tensors (1096 floats, 128 threads)
    for (int i = tid; i < GDIM * EEXP; i += 128) GW2S[i] = gW2[i];
    if (tid < EEXP)  GB2S[tid] = gb2[tid];
    if (tid < H2DIM) OWS[tid]  = oW[tid];

    const int row0 = blockIdx.x * (16 * NWAVES) + wave * 16;

    // ---- stage 1: load x tile, apply input BN, store bf16 to LDS ----
    for (int i = lane; i < (16 * DDIM) / 4; i += 32) {
        int idx = i * 4;
        int r   = idx >> 8;        // /256
        int cc  = idx & 255;
        const float4 xv = *(const float4*)(x + (size_t)(row0 + r) * DDIM + cc);
        const float4 sv = *(const float4*)(in_s + cc);
        const float4 tv = *(const float4*)(in_t + cc);
        unsigned short* d = XN + r * XN_LD + cc;
        d[0] = f2bf(xv.x * sv.x + tv.x);
        d[1] = f2bf(xv.y * sv.y + tv.y);
        d[2] = f2bf(xv.z * sv.z + tv.z);
        d[3] = f2bf(xv.w * sv.w + tv.w);
    }
    __syncthreads();   // publish GW2S/GB2S/OWS (uniform control flow)

    // ---- stage 2: gating layer 1 (256->128, BN folded, ReLU) into H1B ----
    gemm_bias_relu_bf16(XN, XN_LD, g1t, DDIM, GDIM, g1b, H1B, H1_LD, mr, kb, half);

    // ---- stage 3: gating layer 2 (128->8) + softmax -> GATES ----
    for (int p = lane; p < 16 * EEXP; p += 32) {
        int r = p >> 3, e = p & 7;
        float s = GB2S[e];
        const unsigned short* gh = H1B + r * H1_LD;
        for (int k = 0; k < GDIM; ++k) s += bf2f(gh[k]) * GW2S[k * EEXP + e];
        GATES[r * 8 + e] = s;
    }
    if (lane < 16) {
        float* g = GATES + lane * 8;
        float mx = g[0];
#pragma unroll
        for (int e = 1; e < 8; ++e) mx = fmaxf(mx, g[e]);
        float sum = 0.f;
#pragma unroll
        for (int e = 0; e < 8; ++e) { float t = __expf(g[e] - mx); g[e] = t; sum += t; }
        float inv = 1.f / sum;
#pragma unroll
        for (int e = 0; e < 8; ++e) g[e] *= inv;
    }

    // ---- stage 4: experts, gate-weighted accumulate into registers ----
    float acc[4][8];
#pragma unroll
    for (int t = 0; t < 4; ++t)
#pragma unroll
        for (int r = 0; r < 8; ++r) acc[t][r] = 0.f;

    for (int e = 0; e < EEXP; ++e) {
        if (e + 1 < EEXP)  // warm L2/L0 for next expert's layer-0 weights
            __builtin_prefetch(e0t + (size_t)(e + 1) * H0DIM * DDIM, 0, 0);

        // layer 0: 256->256
        gemm_bias_relu_bf16(XN, XN_LD, e0t + (size_t)e * H0DIM * DDIM, DDIM, H0DIM,
                            e0b + e * H0DIM, H0B, H0_LD, mr, kb, half);
        // layer 1: 256->128
        gemm_bias_relu_bf16(H0B, H0_LD, e1t + (size_t)e * H1DIM * H0DIM, H0DIM, H1DIM,
                            e1b + e * H1DIM, H1B, H1_LD, mr, kb, half);
        // layer 2: 128->64, fused gate-weighted accumulate
        const unsigned short* arow = H1B + mr * H1_LD + kb;
        for (int nt = 0; nt < 4; ++nt) {
            const int ncol = nt * 16 + mr;
            const unsigned short* wrow =
                e2t + (size_t)e * H2DIM * H1DIM + (size_t)ncol * H1DIM + kb;
            v8f c = {};
            for (int k0 = 0; k0 < H1DIM; k0 += 32) {
                v16bf a = frag_ld(arow + k0);
                v16bf b = frag_ld(wrow + k0);
                c = __builtin_amdgcn_wmma_f32_16x16x32_bf16(false, a, false, b,
                                                            (short)0, c, false, false);
            }
            const float bv = e2b[e * H2DIM + ncol];
#pragma unroll
            for (int r = 0; r < 8; ++r) {
                int m = half ? (r + 8) : r;
                float v = fmaxf(c[r] + bv, 0.f);
                acc[nt][r] += GATES[m * 8 + e] * v;
            }
        }
    }

    // ---- stage 5: transpose acc through LDS, final 64->1 dot ----
#pragma unroll
    for (int nt = 0; nt < 4; ++nt)
#pragma unroll
        for (int r = 0; r < 8; ++r) {
            int m = half ? (r + 8) : r;
            RED[m * RED_LD + nt * 16 + mr] = acc[nt][r];
        }
    if (lane < 16) {
        const float* rr = RED + lane * RED_LD;
        float s = ob[0];
        for (int n = 0; n < H2DIM; ++n) s += rr[n] * OWS[n];
        out[row0 + lane] = s;
    }
}

// ---------------- host launcher ----------------
extern "C" void kernel_launch(void* const* d_in, const int* in_sizes, int n_in,
                              void* d_out, int out_size, void* d_ws, size_t ws_size,
                              hipStream_t stream) {
    (void)in_sizes; (void)n_in; (void)out_size; (void)ws_size;
    const float* x    = (const float*)d_in[0];
    const float* in_g = (const float*)d_in[1];
    const float* in_b = (const float*)d_in[2];
    const float* in_m = (const float*)d_in[3];
    const float* in_v = (const float*)d_in[4];
    const float* gW1  = (const float*)d_in[5];
    const float* gb1  = (const float*)d_in[6];
    const float* g_g  = (const float*)d_in[7];
    const float* g_b  = (const float*)d_in[8];
    const float* g_m  = (const float*)d_in[9];
    const float* g_v  = (const float*)d_in[10];
    const float* gW2  = (const float*)d_in[11];
    const float* gb2  = (const float*)d_in[12];
    const float* eW0  = (const float*)d_in[13];
    const float* eb0  = (const float*)d_in[14];
    const float* e0g  = (const float*)d_in[15];
    const float* e0bb = (const float*)d_in[16];
    const float* e0m  = (const float*)d_in[17];
    const float* e0v  = (const float*)d_in[18];
    const float* eW1  = (const float*)d_in[19];
    const float* eb1  = (const float*)d_in[20];
    const float* e1g  = (const float*)d_in[21];
    const float* e1bb = (const float*)d_in[22];
    const float* e1m  = (const float*)d_in[23];
    const float* e1v  = (const float*)d_in[24];
    const float* eW2  = (const float*)d_in[25];
    const float* eb2  = (const float*)d_in[26];
    const float* e2g  = (const float*)d_in[27];
    const float* e2bb = (const float*)d_in[28];
    const float* e2m  = (const float*)d_in[29];
    const float* e2v  = (const float*)d_in[30];
    const float* oW   = (const float*)d_in[31];
    const float* ob   = (const float*)d_in[32];
    float* out = (float*)d_out;

    // workspace layout (bytes)
    char* ws = (char*)d_ws;
    float*          ws_in_s = (float*)(ws + 0);
    float*          ws_in_t = (float*)(ws + 1024);
    float*          ws_g1b  = (float*)(ws + 2048);
    float*          ws_e0b  = (float*)(ws + 2560);
    float*          ws_e1b  = (float*)(ws + 10752);
    float*          ws_e2b  = (float*)(ws + 14848);
    unsigned short* ws_g1t  = (unsigned short*)(ws + 16896);
    unsigned short* ws_e0t  = (unsigned short*)(ws + 82432);
    unsigned short* ws_e1t  = (unsigned short*)(ws + 1131008);
    unsigned short* ws_e2t  = (unsigned short*)(ws + 1655296);

    prep_in_bn<<<1, 256, 0, stream>>>(in_g, in_b, in_m, in_v, ws_in_s, ws_in_t);

    {   // gating layer 1: [256,128], E=1
        int total = DDIM * GDIM;
        prep_fold<<<(total + 255) / 256, 256, 0, stream>>>(
            gW1, gb1, g_g, g_b, g_m, g_v, ws_g1t, ws_g1b, DDIM, GDIM, total);
    }
    {   // experts layer 0: [8,256,256]
        int total = EEXP * DDIM * H0DIM;
        prep_fold<<<(total + 255) / 256, 256, 0, stream>>>(
            eW0, eb0, e0g, e0bb, e0m, e0v, ws_e0t, ws_e0b, DDIM, H0DIM, total);
    }
    {   // experts layer 1: [8,256,128]
        int total = EEXP * H0DIM * H1DIM;
        prep_fold<<<(total + 255) / 256, 256, 0, stream>>>(
            eW1, eb1, e1g, e1bb, e1m, e1v, ws_e1t, ws_e1b, H0DIM, H1DIM, total);
    }
    {   // experts layer 2: [8,128,64]
        int total = EEXP * H1DIM * H2DIM;
        prep_fold<<<(total + 255) / 256, 256, 0, stream>>>(
            eW2, eb2, e2g, e2bb, e2m, e2v, ws_e2t, ws_e2b, H1DIM, H2DIM, total);
    }

    const int rows_per_block = 16 * NWAVES;   // 64
    dim3 grid(BATCH / rows_per_block);        // 1024
    moe_fused<<<grid, 128, SMEM_BYTES, stream>>>(
        x, ws_in_s, ws_in_t, ws_g1t, ws_g1b, gW2, gb2,
        ws_e0t, ws_e0b, ws_e1t, ws_e1b, ws_e2t, ws_e2b,
        oW, ob, out);
}
